// GCN2_21242908246487
// MI455X (gfx1250) — matrix operand order repaired and checked by reading, not compiled, not run on hardware
//
#include <hip/hip_runtime.h>
#include <hip/hip_bf16.h>
#include <cstdint>

// ---------------------------------------------------------------------------
// GCN pipeline for MI455X (gfx1250), bf16 WMMA path.
//
// Roofline: gc1_w is 256MB f32 -> HBM floor ~11.5us @ 23.3TB/s; GEMM1 compute
// ~5-6us on bf16 WMMA -> memory bound. GEMM1 uses LDS double buffering:
// B panels staged by the Tensor Data Mover (tensor_load_to_lds, TENSORcnt),
// A panels staged cooperatively (batched loads, f32->bf16 in-register) off
// the WMMA critical path. Split-K with deterministic f32 partials.
// ---------------------------------------------------------------------------

typedef __attribute__((ext_vector_type(16))) __bf16 v16bf;
typedef __attribute__((ext_vector_type(8)))  __bf16 v8bf;
typedef __attribute__((ext_vector_type(8)))  float  v8f;
typedef __attribute__((ext_vector_type(4)))  unsigned int u32x4;
typedef __attribute__((ext_vector_type(8)))  int i32x8;
typedef __attribute__((ext_vector_type(4)))  int i32x4;

#define MT 13  // 208 rows / 16

#if defined(__has_builtin)
#if __has_builtin(__builtin_amdgcn_tensor_load_to_lds) && __has_builtin(__builtin_amdgcn_s_wait_tensorcnt)
#define HAVE_TDM 1
#endif
#endif
#ifndef HAVE_TDM
#define HAVE_TDM 0
#endif

#if HAVE_TDM
// Issue a TDM 2D tile load (f32 elements) global->LDS. Wave-level op
// (EXEC ignored); descriptors per CDNA5 ISA ch.8 (D# groups 0/1).
__device__ inline void tdm_load_2d_f32(unsigned int lds_addr, const void* gaddr,
                                       unsigned int tile_w, unsigned int tile_h,
                                       unsigned int tensor_w, unsigned int tensor_h,
                                       unsigned int stride_elems) {
  uint64_t ga = (uint64_t)(uintptr_t)gaddr;
  u32x4 g0;
  g0[0] = 1u;                                            // count=1, user mode
  g0[1] = lds_addr;                                      // LDS byte address
  g0[2] = (unsigned int)ga;                              // global_addr[31:0]
  g0[3] = (unsigned int)((ga >> 32) & 0x1FFFFFFu) | (2u << 30);  // addr[56:32] | type=2
  i32x8 g1;
  g1[0] = (int)(2u << 16);                               // data_size = 4 bytes
  g1[1] = (int)((tensor_w & 0xFFFFu) << 16);             // tensor_dim0[15:0] @ bits 63:48
  g1[2] = (int)(((tensor_w >> 16) & 0xFFFFu) | ((tensor_h & 0xFFFFu) << 16));
  g1[3] = (int)(((tensor_h >> 16) & 0xFFFFu) | ((tile_w & 0xFFFFu) << 16));
  g1[4] = (int)(tile_h & 0xFFFFu);                       // tile_dim1 ; tile_dim2 = 0
  g1[5] = (int)stride_elems;                             // tensor_dim0_stride[31:0]
  g1[6] = 0;                                             // stride[47:32], dim1_stride lo
  g1[7] = 0;
  i32x4 g2 = {0, 0, 0, 0};
  i32x4 g3 = {0, 0, 0, 0};
#if __clang_major__ >= 23
  i32x8 g4 = {0, 0, 0, 0, 0, 0, 0, 0};
  __builtin_amdgcn_tensor_load_to_lds(g0, g1, g2, g3, g4, 0);
#else
  __builtin_amdgcn_tensor_load_to_lds(g0, g1, g2, g3, 0);
#endif
}
#endif

// ---------------- fragment loaders (lane k = 16*hi + s; load_a_f32 uses ISA table order) -------------
__device__ inline v16bf load_a_f32(const float* __restrict__ A, int lda,
                                   int row, int k, int hi) {
  const float* p = A + (size_t)row * lda + k + hi * 8;
  float4 a0 = *(const float4*)(p + 0);
  float4 a1 = *(const float4*)(p + 4);
  float4 b0 = *(const float4*)(p + 16);
  float4 b1 = *(const float4*)(p + 20);
  v16bf r;
  r[0]=(__bf16)a0.x; r[1]=(__bf16)a0.y; r[2]=(__bf16)a0.z; r[3]=(__bf16)a0.w;
  r[4]=(__bf16)a1.x; r[5]=(__bf16)a1.y; r[6]=(__bf16)a1.z; r[7]=(__bf16)a1.w;
  r[8]=(__bf16)b0.x; r[9]=(__bf16)b0.y; r[10]=(__bf16)b0.z; r[11]=(__bf16)b0.w;
  r[12]=(__bf16)b1.x; r[13]=(__bf16)b1.y; r[14]=(__bf16)b1.z; r[15]=(__bf16)b1.w;
  return r;
}

__device__ inline v16bf load_a_bf16(const __bf16* __restrict__ A, int lda,
                                    int row, int k, int hi) {
  const __bf16* p = A + (size_t)row * lda + k + hi * 16;
  v8bf lo = *(const v8bf*)(p);
  v8bf hiV = *(const v8bf*)(p + 8);
  v16bf r;
#pragma unroll
  for (int j = 0; j < 8; ++j) { r[j] = lo[j]; r[8 + j] = hiV[j]; }
  return r;
}

__device__ inline v16bf load_b_f32(const float* __restrict__ B, int ldb,
                                   int k, int col, int hi) {
  const float* p = B + (size_t)(k + hi * 16) * ldb + col;
  __builtin_prefetch(p + 32 * (size_t)ldb, 0, 1);
  v16bf r;
#pragma unroll
  for (int j = 0; j < 8; ++j) {
    r[2 * j]     = (__bf16)p[0];
    r[2 * j + 1] = (__bf16)p[ldb];
    p += 2 * (size_t)ldb;
  }
  return r;
}

__device__ inline v16bf load_b_bf16(const __bf16* __restrict__ B, int ldb,
                                    int k, int col, int hi) {
  const __bf16* p = B + (size_t)(k + hi * 16) * ldb + col;
  __builtin_prefetch(p + 32 * (size_t)ldb, 0, 1);
  v16bf r;
#pragma unroll
  for (int j = 0; j < 8; ++j) {
    r[2 * j]     = p[0];
    r[2 * j + 1] = p[ldb];
    p += 2 * (size_t)ldb;
  }
  return r;
}

// ---------------------------------------------------------------------------
// GEMM1: support1 = x[208x8192] @ gc1_w[8192x8192], split-K f32 partials.
// 8 waves/block, 128-col panel, all 13 M tiles per wave, LDS double buffer.
// ---------------------------------------------------------------------------
#define G1_LD 8192

__global__ __launch_bounds__(256)
void gemm1_tdm_kernel(const float* __restrict__ A, const float* __restrict__ B,
                      float* __restrict__ Cpart, int kChunk) {
  __shared__ __bf16 ldsA[2][208 * 32];   // A panel, bf16   (2 x 13KB)
  __shared__ float  ldsB[2][32 * 128];   // B panel, f32    (2 x 16KB)

  const int tid   = threadIdx.x;
  const int lane  = tid & 31;
  const int wv    = tid >> 5;
  const int hi    = lane >> 4;
  const int lcol  = lane & 15;
  const int nbase = blockIdx.x * 128;
  const int k0    = blockIdx.y * kChunk;
  const int ksteps = kChunk >> 5;

  v8f acc[MT];
#pragma unroll
  for (int m = 0; m < MT; ++m) acc[m] = v8f{0.f,0.f,0.f,0.f,0.f,0.f,0.f,0.f};

  // Batched A staging: 208x16 float2 = 13 per thread (exact). Issue all
  // global loads first (MLP), then convert + ds_store.
  auto stageA = [&](int s, int buf) {
    const int k = k0 + s * 32;
    float2 tmp[13];
#pragma unroll
    for (int i = 0; i < 13; ++i) {
      const int p = tid + i * 256;
      const int r = p >> 4;
      const int c = (p & 15) << 1;
      tmp[i] = *(const float2*)(A + (size_t)r * G1_LD + k + c);
    }
#pragma unroll
    for (int i = 0; i < 13; ++i) {
      const int p = tid + i * 256;
      const int r = p >> 4;
      const int c = (p & 15) << 1;
      ldsA[buf][r * 32 + c]     = (__bf16)tmp[i].x;
      ldsA[buf][r * 32 + c + 1] = (__bf16)tmp[i].y;
    }
  };
  auto issueB = [&](int s, int buf) {
    const int k = k0 + s * 32;
#if HAVE_TDM
    if (tid < 32) {  // one wave issues the DMA; TDM ignores EXEC
      unsigned int laddr = (unsigned int)(uintptr_t)&ldsB[buf][0];
      tdm_load_2d_f32(laddr, B + (size_t)k * G1_LD + nbase,
                      /*tile_w*/128, /*tile_h*/32,
                      /*tensor_w*/G1_LD, /*tensor_h*/8192, /*stride*/G1_LD);
    }
#else
    for (int p = tid; p < 32 * 128; p += 256) {
      const int r = p >> 7, c = p & 127;
      ldsB[buf][p] = B[(size_t)(k + r) * G1_LD + nbase + c];
    }
#endif
  };

  stageA(0, 0);
  issueB(0, 0);

  for (int s = 0; s < ksteps; ++s) {
    const int buf = s & 1;
    if (s + 1 < ksteps) {          // async B for next step, then wait stage s
      issueB(s + 1, buf ^ 1);
#if HAVE_TDM
      __builtin_amdgcn_s_wait_tensorcnt(1);
#endif
    } else {
#if HAVE_TDM
      __builtin_amdgcn_s_wait_tensorcnt(0);
#endif
    }
    __syncthreads();

    // B fragment from LDS (f32 -> bf16): lane = col, K packed 2/VGPR.
    v16bf bfrag;
#pragma unroll
    for (int j = 0; j < 8; ++j) {
      bfrag[2 * j]     = (__bf16)ldsB[buf][(hi * 16 + 2 * j) * 128 + wv * 16 + lcol];
      bfrag[2 * j + 1] = (__bf16)ldsB[buf][(hi * 16 + 2 * j + 1) * 128 + wv * 16 + lcol];
    }
#pragma unroll
    for (int m = 0; m < MT; ++m) {
      const __bf16* pa = &ldsA[buf][(m * 16 + lcol) * 32 + hi * 16];
      v8bf lo = *(const v8bf*)pa;         // ds_load_b128
      v8bf hh = *(const v8bf*)(pa + 8);
      v16bf afrag;
#pragma unroll
      for (int j = 0; j < 8; ++j) { afrag[j] = lo[j]; afrag[8 + j] = hh[j]; }
      acc[m] = __builtin_amdgcn_wmma_f32_16x16x32_bf16(
          false, afrag, false, bfrag, (short)0, acc[m], false, false);
    }
    // A staging for next step AFTER compute: its ds_stores retire before the
    // end barrier instead of stalling this step's wmma stream.
    if (s + 1 < ksteps) stageA(s + 1, buf ^ 1);
    __syncthreads();
  }

  float* C = Cpart + (size_t)blockIdx.y * (size_t)208 * 8192;
  const int col = nbase + wv * 16 + lcol;
  const int rowoff = hi * 8;
#pragma unroll
  for (int m = 0; m < MT; ++m) {
    const int rbase = m * 16 + rowoff;
#pragma unroll
    for (int r = 0; r < 8; ++r)
      C[(size_t)(rbase + r) * 8192 + col] = acc[m][r];
  }
}

// ---------------- generic direct-load WMMA GEMM (light stages) --------------
// AMODE/BMODE: 0 = f32 source, 1 = bf16 source.
// EPI: 0 = f32 partial @ C + blockIdx.y*partStride, 1 = bias+relu bf16,
//      2 = bias+relu f32.
template <int AMODE, int BMODE, int EPI>
__global__ __launch_bounds__(32)
void gemm_wmma_kernel(const void* __restrict__ Av, int lda,
                      const void* __restrict__ Bv, int ldb,
                      const float* __restrict__ bias,
                      void* __restrict__ Cv, int ldc,
                      size_t partStride, int kChunk) {
  const int lane  = threadIdx.x & 31;
  const int hi    = lane >> 4;
  const int lcol  = lane & 15;
  const int nbase = blockIdx.x * 16;
  const int col   = nbase + lcol;
  const int k0    = blockIdx.y * kChunk;
  const int ksteps = kChunk >> 5;

  v8f acc[MT];
#pragma unroll
  for (int m = 0; m < MT; ++m) acc[m] = v8f{0.f,0.f,0.f,0.f,0.f,0.f,0.f,0.f};

  for (int s = 0; s < ksteps; ++s) {
    const int k = k0 + s * 32;
    v16bf bfrag;
    if constexpr (BMODE == 0) bfrag = load_b_f32((const float*)Bv, ldb, k, col, hi);
    else                      bfrag = load_b_bf16((const __bf16*)Bv, ldb, k, col, hi);
#pragma unroll
    for (int m = 0; m < MT; ++m) {
      const int row = m * 16 + lcol;
      v16bf afrag;
      if constexpr (AMODE == 0) afrag = load_a_f32((const float*)Av, lda, row, k, hi);
      else                      afrag = load_a_bf16((const __bf16*)Av, lda, row, k, hi);
      acc[m] = __builtin_amdgcn_wmma_f32_16x16x32_bf16(
          false, afrag, false, bfrag, (short)0, acc[m], false, false);
    }
  }

  const int rowoff = hi * 8;
  if constexpr (EPI == 0) {
    float* C = (float*)Cv + (size_t)blockIdx.y * partStride;
#pragma unroll
    for (int m = 0; m < MT; ++m) {
      const int rbase = m * 16 + rowoff;
#pragma unroll
      for (int r = 0; r < 8; ++r)
        C[(size_t)(rbase + r) * ldc + col] = acc[m][r];
    }
  } else if constexpr (EPI == 1) {
    __bf16* C = (__bf16*)Cv;
    const float bv = bias[col];
#pragma unroll
    for (int m = 0; m < MT; ++m) {
      const int rbase = m * 16 + rowoff;
#pragma unroll
      for (int r = 0; r < 8; ++r) {
        float v = acc[m][r] + bv;
        v = v > 0.f ? v : 0.f;
        C[(size_t)(rbase + r) * ldc + col] = (__bf16)v;
      }
    }
  } else {
    float* C = (float*)Cv;
    const float bv = bias[col];
#pragma unroll
    for (int m = 0; m < MT; ++m) {
      const int rbase = m * 16 + rowoff;
#pragma unroll
      for (int r = 0; r < 8; ++r) {
        float v = acc[m][r] + bv;
        v = v > 0.f ? v : 0.f;
        C[(size_t)(rbase + r) * ldc + col] = v;
      }
    }
  }
}

// ---------------- helpers ---------------------------------------------------
__global__ void adj_to_bf16_pad(const float* __restrict__ adj,
                                __bf16* __restrict__ out) {
  int idx = blockIdx.x * blockDim.x + threadIdx.x;
  if (idx >= 208 * 224) return;
  int r = idx / 224, c = idx % 224;
  out[idx] = (__bf16)(c < 208 ? adj[r * 208 + c] : 0.f);
}

template <int NPART>
__global__ void reduce_parts_bf16(const float* __restrict__ parts, size_t partStride,
                                  __bf16* __restrict__ out,
                                  int rows, int cols, int padRows) {
  size_t idx = (size_t)blockIdx.x * blockDim.x + threadIdx.x;
  size_t total = (size_t)padRows * cols;
  if (idx >= total) return;
  int r = (int)(idx / cols);
  int c = (int)(idx % cols);
  float s = 0.f;
  if (r < rows) {
    size_t o = (size_t)r * cols + c;
#pragma unroll
    for (int p = 0; p < NPART; ++p) s += parts[(size_t)p * partStride + o];
  }
  out[idx] = (__bf16)s;
}

__global__ void fc1_kernel(const float* __restrict__ w, const float* __restrict__ b,
                           const float* __restrict__ h2, float* __restrict__ out) {
  __shared__ float red[256];
  const int row = blockIdx.x, tid = threadIdx.x;
  float s = 0.f;
  for (int i = tid; i < 13312; i += 256) s += w[(size_t)row * 13312 + i] * h2[i];
  red[tid] = s;
  __syncthreads();
  for (int off = 128; off > 0; off >>= 1) {
    if (tid < off) red[tid] += red[tid + off];
    __syncthreads();
  }
  if (tid == 0) {
    float v = red[0] + b[row];
    out[row] = v > 0.f ? v : 0.f;
  }
}

__global__ void head_kernel(const float* __restrict__ w2, const float* __restrict__ b2,
                            const float* __restrict__ w3, const float* __restrict__ b3,
                            const float* __restrict__ v1, float* __restrict__ out) {
  __shared__ float buf[32];
  const int l = threadIdx.x;
  float s = 0.f;
  for (int i = 0; i < 128; ++i) s += w2[l * 128 + i] * v1[i];
  s += b2[l];
  s = s > 0.f ? s : 0.f;
  buf[l] = s;
  __syncthreads();
  if (l == 0) {
    float t = 0.f;
    for (int i = 0; i < 32; ++i) t += w3[i] * buf[i];
    t += b3[0];
    out[0] = 1.f / (1.f + expf(-t));
  }
}

// ---------------- workspace layout ------------------------------------------
static constexpr size_t AL(size_t x) { return (x + 255) & ~(size_t)255; }
static constexpr size_t OFF_ADJ = 0;
static constexpr size_t SZ_ADJ  = (size_t)208 * 224 * 2;
static constexpr size_t OFF_S1P = AL(OFF_ADJ + SZ_ADJ);
static constexpr size_t S1_PART_ELEMS = (size_t)208 * 8192;
static constexpr size_t SZ_S1P  = 4 * S1_PART_ELEMS * 4;
static constexpr size_t OFF_S1  = AL(OFF_S1P + SZ_S1P);
static constexpr size_t SZ_S1   = (size_t)224 * 8192 * 2;
static constexpr size_t OFF_H1  = AL(OFF_S1 + SZ_S1);
static constexpr size_t SZ_H1   = (size_t)208 * 8192 * 2;
static constexpr size_t OFF_S2P = AL(OFF_H1 + SZ_H1);
static constexpr size_t S2_PART_ELEMS = (size_t)208 * 64;
static constexpr size_t SZ_S2P  = 16 * S2_PART_ELEMS * 4;
static constexpr size_t OFF_S2  = AL(OFF_S2P + SZ_S2P);
static constexpr size_t SZ_S2   = (size_t)224 * 64 * 2;
static constexpr size_t OFF_H2  = AL(OFF_S2 + SZ_S2);
static constexpr size_t SZ_H2   = (size_t)208 * 64 * 4;
static constexpr size_t OFF_FC1 = AL(OFF_H2 + SZ_H2);

extern "C" void kernel_launch(void* const* d_in, const int* in_sizes, int n_in,
                              void* d_out, int out_size, void* d_ws, size_t ws_size,
                              hipStream_t stream) {
  const float* x     = (const float*)d_in[0];
  const float* adj   = (const float*)d_in[1];
  const float* gc1_w = (const float*)d_in[2];
  const float* gc1_b = (const float*)d_in[3];
  const float* gc2_w = (const float*)d_in[4];
  const float* gc2_b = (const float*)d_in[5];
  const float* fc1_w = (const float*)d_in[6];
  const float* fc1_b = (const float*)d_in[7];
  const float* fc2_w = (const float*)d_in[8];
  const float* fc2_b = (const float*)d_in[9];
  const float* fc3_w = (const float*)d_in[10];
  const float* fc3_b = (const float*)d_in[11];

  char* ws = (char*)d_ws;
  __bf16* adjb = (__bf16*)(ws + OFF_ADJ);
  float*  s1p  = (float*)(ws + OFF_S1P);
  __bf16* s1   = (__bf16*)(ws + OFF_S1);
  __bf16* h1   = (__bf16*)(ws + OFF_H1);
  float*  s2p  = (float*)(ws + OFF_S2P);
  __bf16* s2   = (__bf16*)(ws + OFF_S2);
  float*  h2   = (float*)(ws + OFF_H2);
  float*  fc1o = (float*)(ws + OFF_FC1);

  // 0) adj -> padded bf16 [208 x 224]
  adj_to_bf16_pad<<<(208 * 224 + 255) / 256, 256, 0, stream>>>(adj, adjb);

  // 1) support1 = x @ gc1_w  (LDS + TDM pipeline, split-K x4)
  gemm1_tdm_kernel<<<dim3(64, 4), 256, 0, stream>>>(x, gc1_w, s1p, 2048);
  reduce_parts_bf16<4><<<(int)((224 * 8192 + 255) / 256), 256, 0, stream>>>(
      s1p, S1_PART_ELEMS, s1, 208, 8192, 224);

  // 2) h1 = relu(adj @ support1 + gc1_b)
  gemm_wmma_kernel<1, 1, 1><<<dim3(512, 1), 32, 0, stream>>>(
      adjb, 224, s1, 8192, gc1_b, h1, 8192, 0, 224);

  // 3) support2 = h1 @ gc2_w  (split-K x16)
  gemm_wmma_kernel<1, 0, 0><<<dim3(4, 16), 32, 0, stream>>>(
      h1, 8192, gc2_w, 64, nullptr, s2p, 64, S2_PART_ELEMS, 512);
  reduce_parts_bf16<16><<<(224 * 64 + 255) / 256, 256, 0, stream>>>(
      s2p, S2_PART_ELEMS, s2, 208, 64, 224);

  // 4) h2 = relu(adj @ support2 + gc2_b)
  gemm_wmma_kernel<1, 1, 2><<<dim3(4, 1), 32, 0, stream>>>(
      adjb, 224, s2, 64, gc2_b, h2, 64, 0, 224);

  // 5) fc1 + relu
  fc1_kernel<<<128, 256, 0, stream>>>(fc1_w, fc1_b, h2, fc1o);

  // 6) fc2 + relu + fc3 + sigmoid
  head_kernel<<<1, 32, 0, stream>>>(fc2_w, fc2_b, fc3_w, fc3_b, fc1o, (float*)d_out);
}
